// AttentionGRU_41025527611762
// MI455X (gfx1250) — compile-verified, matrix operand order checked by reading, not run on hardware
//
#include <hip/hip_runtime.h>
#include <hip/hip_bf16.h>
#include <cstddef>
#include <cstdint>

typedef _Float16 half_t;
typedef __attribute__((ext_vector_type(8)))  _Float16 v8h;
typedef __attribute__((ext_vector_type(16))) _Float16 v16h;
typedef __attribute__((ext_vector_type(8)))  float    v8f;

// ---------------- problem constants ----------------
constexpr int kB   = 512;
constexpr int kT   = 100;
constexpr int kEMB = 128;
constexpr int kH   = 256;
constexpr int kOUT = 128;
constexpr int kECFP= 1024;
constexpr int kFX  = 78;
constexpr int kHEADS = 2;
constexpr int kNPG = 40;
constexpr int kDEG = 8;
constexpr int kN   = kB * kNPG;       // 20480 nodes
constexpr int kE   = kN * kDEG;       // 163840 input edges
constexpr int kET  = kE + kN;         // + self loops
constexpr int kH3  = 3 * kH;          // 768
constexpr int kH2  = 2 * kH;          // 512
constexpr int kBT  = kB * kT;         // 51200
constexpr int kGC  = kHEADS * kFX;    // 156
constexpr int kGCp = 160;             // padded N for GAT
constexpr int kFXp = 96;              // padded K for GAT layer 1

// ---------------- small device helpers ----------------
__device__ __forceinline__ float sigmf(float x){ return 1.f/(1.f+__expf(-x)); }

__device__ void atomicMaxF(float* addr, float val){
  unsigned int* ua = (unsigned int*)addr;
  unsigned int old = __float_as_uint(*addr);
  while (__uint_as_float(old) < val){
    unsigned int assumed = old;
    old = atomicCAS(ua, assumed, __float_as_uint(val));
    if (old == assumed) break;
  }
}

// ---------------- utility kernels ----------------
__global__ void fill_f32_k(float* p, float v, int n){
  int i = blockIdx.x*blockDim.x + threadIdx.x;
  if (i < n) p[i] = v;
}
__global__ void fill_f16_k(half_t* p, int n){
  int i = blockIdx.x*blockDim.x + threadIdx.x;
  if (i < n) p[i] = (half_t)0.f;
}
__global__ void cvt_f16_k(const float* __restrict__ s, half_t* __restrict__ d, int n){
  int i = blockIdx.x*blockDim.x + threadIdx.x;
  if (i < n) d[i] = (half_t)s[i];
}
// dst (row-major, Cd cols): dst[r][c] = (r<R && c<C) ? src[r][c] : 0
__global__ void cvt_pad_f16_k(const float* __restrict__ s, int R, int C,
                              half_t* __restrict__ d, int Cd, int total){
  int i = blockIdx.x*blockDim.x + threadIdx.x;
  if (i >= total) return;
  int r = i / Cd, c = i % Cd;
  d[i] = (r < R && c < C) ? (half_t)s[(size_t)r*C + c] : (half_t)0.f;
}
// dst[i][j] = src[j][i] ; src is (R,C); dst is (*,Cd); valid i<C, j<R
__global__ void transpose_pad_f16_k(const float* __restrict__ s, int R, int C,
                                    half_t* __restrict__ d, int Cd, int total){
  int idx = blockIdx.x*blockDim.x + threadIdx.x;
  if (idx >= total) return;
  int i = idx / Cd, j = idx % Cd;
  d[idx] = (i < C && j < R) ? (half_t)s[(size_t)j*C + i] : (half_t)0.f;
}

// ---------------- embedding + positional encoding ----------------
__global__ void embed_k(const int* __restrict__ ids, const float* __restrict__ table,
                        half_t* __restrict__ out){
  int idx = blockIdx.x*blockDim.x + threadIdx.x;
  if (idx >= kBT*kEMB) return;
  int bt = idx / kEMB, e = idx % kEMB;
  int t = bt % kT;
  float v = table[(size_t)ids[bt]*kEMB + e] * 11.3137084989847604f;   // sqrt(128)
  int hi = e >> 1;
  float div = __expf(-(float)(2*hi) * (9.21034037197618f / 128.f));   // ln(1e4)
  float ang = (float)t * div;
  v += (e & 1) ? __cosf(ang) : __sinf(ang);
  out[idx] = (half_t)v;
}

// ---------------- WMMA GEMM: C(M,N) = A(M,K)f16 * W(N,K)f16^T + bias ----------------
// A row-major (lda), W row-major (ldw); both fragments load with the 16-bit
// A-matrix lane layout from the CDNA5 ISA (W rows == columns of B).
__global__ __launch_bounds__(256)
void gemm_wmma_k(const half_t* __restrict__ A, int lda,
                 const half_t* __restrict__ W, int ldw,
                 float* __restrict__ C, int ldc,
                 half_t* __restrict__ Ch, int ldch,
                 const float* __restrict__ bias,
                 int M, int Nn, int K, int act){
  int wave = threadIdx.x >> 5;          // wave32
  int lane = threadIdx.x & 31;
  int m0 = blockIdx.x * 16;
  int n0 = (blockIdx.y * 8 + wave) * 16;
  if (n0 >= Nn) return;                 // wave-uniform: EXEC stays all-ones
  int hsel = lane >> 4;                 // 0: K {0..7,16..23}, 1: K {8..15,24..31}
  int r = lane & 15;
  const half_t* arow = A + (size_t)(m0 + r) * lda;
  const half_t* wrow = W + (size_t)(n0 + r) * ldw;
  v8f acc = {};
  for (int k0 = 0; k0 < K; k0 += 32){
    int kb = k0 + hsel * 8;
    v8h a0 = *(const v8h*)(arow + kb);
    v8h a1 = *(const v8h*)(arow + kb + 16);
    v8h b0 = *(const v8h*)(wrow + kb);
    v8h b1 = *(const v8h*)(wrow + kb + 16);
    v16h av = __builtin_shufflevector(a0, a1, 0,1,2,3,4,5,6,7,8,9,10,11,12,13,14,15);
    v16h bv = __builtin_shufflevector(b0, b1, 0,1,2,3,4,5,6,7,8,9,10,11,12,13,14,15);
    acc = __builtin_amdgcn_wmma_f32_16x16x32_f16(false, av, false, bv,
                                                 (short)0, acc, false, false);
  }
  int col = n0 + r;
  if (col >= Nn) return;
  float bv2 = bias ? bias[col] : 0.f;
  int rb = m0 + hsel * 8;               // C layout: VGPR i -> row rb+i, col = n0+(lane&15)
  #pragma unroll
  for (int i = 0; i < 8; i++){
    float v = acc[i] + bv2;
    if (act == 1) v = tanhf(v);
    else if (act == 2) v = fmaxf(v, 0.f);
    size_t o = (size_t)(rb + i) * ldc + col;
    C[o] = v;
    if (Ch) Ch[(size_t)(rb + i) * ldch + col] = (half_t)v;
  }
}

// ---------------- GRU gate kernels ----------------
__global__ void gru_gate_k(const float* __restrict__ xw, const float* __restrict__ gh,
                           float* __restrict__ h, half_t* __restrict__ hh,
                           float* __restrict__ S, half_t* __restrict__ Sh,
                           int t, int dirOff){
  int i = blockIdx.x*blockDim.x + threadIdx.x;
  if (i >= kB*kH) return;
  int b = i / kH, j = i % kH;
  const float* xr = xw + ((size_t)b*kT + t) * kH3;
  const float* gr = gh + (size_t)b * kH3;
  float r  = sigmf(xr[j] + gr[j]);
  float z  = sigmf(xr[kH + j] + gr[kH + j]);
  float nn = tanhf(xr[2*kH + j] + r * gr[2*kH + j]);
  float hv = (1.f - z) * nn + z * h[i];
  h[i]  = hv;
  hh[i] = (half_t)hv;
  size_t o = ((size_t)b*kT + t) * kH2 + dirOff + j;
  S[o] = hv;
  Sh[o] = (half_t)hv;
}
// single-step GRU with h0 = 0 (gh = bhh)
__global__ void gru_single_k(const float* __restrict__ xw, const float* __restrict__ bhh,
                             float* __restrict__ S, half_t* __restrict__ Sh, int dirOff){
  int i = blockIdx.x*blockDim.x + threadIdx.x;
  if (i >= kB*kH) return;
  int b = i / kH, j = i % kH;
  const float* xr = xw + (size_t)b * kH3;
  float r  = sigmf(xr[j] + bhh[j]);
  float z  = sigmf(xr[kH + j] + bhh[kH + j]);
  float nn = tanhf(xr[2*kH + j] + r * bhh[2*kH + j]);
  float hv = (1.f - z) * nn;
  size_t o = (size_t)b * kH2 + dirOff + j;
  S[o] = hv;
  if (Sh) Sh[o] = (half_t)hv;
}

// ---------------- layernorm (one block per row) ----------------
__global__ void layernorm_k(const float* __restrict__ X, const float* __restrict__ g,
                            const float* __restrict__ b, float* __restrict__ Y,
                            half_t* __restrict__ Yh, int C){
  int row = blockIdx.x, tid = threadIdx.x;
  const float* xr = X + (size_t)row * C;
  float s = 0.f, s2 = 0.f;
  for (int c = tid; c < C; c += blockDim.x){ float v = xr[c]; s += v; s2 += v*v; }
  __shared__ float sh[256], sh2[256];
  sh[tid] = s; sh2[tid] = s2; __syncthreads();
  for (int st = 128; st > 0; st >>= 1){
    if (tid < st){ sh[tid] += sh[tid+st]; sh2[tid] += sh2[tid+st]; }
    __syncthreads();
  }
  float mean = sh[0] / C;
  float var  = sh2[0] / C - mean*mean;
  float inv  = rsqrtf(var + 1e-5f);
  for (int c = tid; c < C; c += blockDim.x){
    float v = (xr[c] - mean) * inv * g[c] + b[c];
    Y[(size_t)row*C + c] = v;
    if (Yh) Yh[(size_t)row*C + c] = (half_t)v;
  }
}

// ---------------- attention kernels ----------------
__global__ void softmax_T_k(float* __restrict__ S){   // in-place over T, layout (B*T, OUT)
  int i = blockIdx.x*blockDim.x + threadIdx.x;
  if (i >= kB*kOUT) return;
  int b = i / kOUT, c = i % kOUT;
  float* base = S + (size_t)b*kT*kOUT + c;
  float m = -1e30f;
  for (int t = 0; t < kT; t++) m = fmaxf(m, base[(size_t)t*kOUT]);
  float s = 0.f;
  for (int t = 0; t < kT; t++) s += __expf(base[(size_t)t*kOUT] - m);
  float inv = 1.f / s;
  for (int t = 0; t < kT; t++){
    size_t o = (size_t)t*kOUT;
    base[o] = __expf(base[o] - m) * inv;
  }
}
__global__ void mul_acc_k(float* __restrict__ acc, const float* __restrict__ a,
                          const float* __restrict__ b, int n){
  int i = blockIdx.x*blockDim.x + threadIdx.x;
  if (i < n) acc[i] += a[i] * b[i];
}
__global__ void combine05_k(float* __restrict__ o, const float* __restrict__ a,
                            const float* __restrict__ b, int n){
  int i = blockIdx.x*blockDim.x + threadIdx.x;
  if (i < n) o[i] = 0.5f * a[i] + b[i];
}
__global__ void ep_combine_k(float* __restrict__ o, const float* __restrict__ a,
                             const float* __restrict__ b, const float* __restrict__ c, int n){
  int i = blockIdx.x*blockDim.x + threadIdx.x;
  if (i < n) o[i] = 0.5f * (a[i] + b[i]) + c[i];
}
// out[b][o] = sum_t resh(B,OUT,T)[b][o][t] * w[t] + bias ; resh is flat reinterpret
__global__ void smi_fc2_k(const float* __restrict__ att, const float* __restrict__ w,
                          const float* __restrict__ b1, float* __restrict__ out){
  int i = blockIdx.x*blockDim.x + threadIdx.x;
  if (i >= kB*kOUT) return;
  int bb = i / kOUT, o = i % kOUT;
  const float* base = att + (size_t)bb * kT * kOUT;   // contiguous 12800 floats / batch
  float s = 0.f;
  for (int t = 0; t < kT; t++) s += base[o*kT + t] * w[t];
  out[i] = s + b1[0];
}

// ---------------- GAT kernels ----------------
__global__ void gat_scores_k(const float* __restrict__ h, int ldh,
                             const float* __restrict__ asrc, const float* __restrict__ adst,
                             float* __restrict__ ssrc, float* __restrict__ sdst){
  int i = blockIdx.x*blockDim.x + threadIdx.x;
  if (i >= kN*kHEADS) return;
  int n = i / kHEADS, hd = i % kHEADS;
  const float* hr = h + (size_t)n*ldh + hd*kFX;
  const float* as = asrc + hd*kFX;
  const float* ad = adst + hd*kFX;
  float s1 = 0.f, s2 = 0.f;
  for (int k = 0; k < kFX; k++){ float v = hr[k]; s1 += v*as[k]; s2 += v*ad[k]; }
  ssrc[i] = s1; sdst[i] = s2;
}
__global__ void gat_edge_max_k(const int* __restrict__ ei, const float* __restrict__ ssrc,
                               const float* __restrict__ sdst,
                               float* __restrict__ ev, float* __restrict__ mx){
  int i = blockIdx.x*blockDim.x + threadIdx.x;
  if (i >= kET*kHEADS) return;
  int e = i >> 1, hd = i & 1;
  int s = (e < kE) ? ei[e]       : (e - kE);
  int d = (e < kE) ? ei[kE + e]  : (e - kE);
  float v = ssrc[s*kHEADS + hd] + sdst[d*kHEADS + hd];
  v = v > 0.f ? v : 0.2f * v;                 // leaky relu 0.2
  ev[i] = v;
  atomicMaxF(&mx[d*kHEADS + hd], v);
}
__global__ void gat_edge_exp_k(const int* __restrict__ ei, float* __restrict__ ev,
                               const float* __restrict__ mx, float* __restrict__ den){
  int i = blockIdx.x*blockDim.x + threadIdx.x;
  if (i >= kET*kHEADS) return;
  int e = i >> 1, hd = i & 1;
  int d = (e < kE) ? ei[kE + e] : (e - kE);
  float a = __expf(ev[i] - mx[d*kHEADS + hd]);
  ev[i] = a;
  atomicAdd(&den[d*kHEADS + hd], a);
}
__global__ void gat_scatter_k(const int* __restrict__ ei, const float* __restrict__ ev,
                              const float* __restrict__ den, const float* __restrict__ h,
                              float* __restrict__ acc){
  int e = blockIdx.x, k = threadIdx.x;
  if (k >= kGC) return;
  int s = (e < kE) ? ei[e]      : (e - kE);
  int d = (e < kE) ? ei[kE + e] : (e - kE);
  int hd = k / kFX;
  float a = ev[(size_t)e*kHEADS + hd] / (den[(size_t)d*kHEADS + hd] + 1e-16f);
  atomicAdd(&acc[(size_t)d*kGC + k], a * h[(size_t)s*kGC + k]);
}
__global__ void gat_bias_relu_k(const float* __restrict__ acc, const float* __restrict__ bias,
                                float* __restrict__ g, half_t* __restrict__ gh){
  int i = blockIdx.x*blockDim.x + threadIdx.x;
  if (i >= kN*kGCp) return;
  int n = i / kGCp, k = i % kGCp;
  float v = 0.f;
  if (k < kGC){
    v = fmaxf(acc[(size_t)n*kGC + k] + bias[k], 0.f);
    g[(size_t)n*kGC + k] = v;
  }
  if (gh) gh[i] = (half_t)v;   // zero padding in padded f16 copy
}
__global__ void gat_pool_k(const int* __restrict__ batch, const float* __restrict__ g,
                           float* __restrict__ ga, float* __restrict__ cnt){
  int i = blockIdx.x*blockDim.x + threadIdx.x;
  if (i >= kN*kGC) return;
  int n = i / kGC, k = i % kGC;
  atomicAdd(&ga[(size_t)batch[n]*kGC + k], g[i]);
  if (k == 0) atomicAdd(&cnt[batch[n]], 1.f);
}
__global__ void gat_div_k(float* __restrict__ ga, const float* __restrict__ cnt){
  int i = blockIdx.x*blockDim.x + threadIdx.x;
  if (i >= kB*kGC) return;
  ga[i] /= cnt[i / kGC];
}

// ---------------- final output ----------------
__global__ void final_k(const float* __restrict__ smiatt, const float* __restrict__ epatt,
                        const float* __restrict__ ga, const float* __restrict__ w,
                        const float* __restrict__ b, float* __restrict__ out){
  int bb = blockIdx.x*blockDim.x + threadIdx.x;
  if (bb >= kB) return;
  float s = 0.f;
  for (int c = 0; c < kOUT; c++) s += smiatt[(size_t)bb*kOUT + c] * w[c];
  for (int c = 0; c < kOUT; c++) s += epatt[(size_t)bb*kOUT + c] * w[kOUT + c];
  for (int k = 0; k < kGC;  k++) s += ga[(size_t)bb*kGC + k] * w[2*kOUT + k];
  out[bb] = s + b[0];
}

// ---------------- host helpers ----------------
static inline void gemm(hipStream_t st, const half_t* A, int lda,
                        const half_t* W, int ldw,
                        float* C, int ldc, half_t* Ch, int ldch,
                        const float* bias, int M, int Nn, int K, int act){
  int tiles = (Nn + 15) / 16;
  dim3 g(M / 16, (tiles + 7) / 8);
  gemm_wmma_k<<<g, 256, 0, st>>>(A, lda, W, ldw, C, ldc, Ch, ldch, bias, M, Nn, K, act);
}

#define L1D(kern, n, ...) kern<<<((int)(n)+255)/256, 256, 0, stream>>>(__VA_ARGS__)

extern "C" void kernel_launch(void* const* d_in, const int* in_sizes, int n_in,
                              void* d_out, int out_size, void* d_ws, size_t ws_size,
                              hipStream_t stream){
  (void)in_sizes; (void)n_in; (void)out_size; (void)ws_size;

  // ---- inputs (setup_inputs dict order; params flattened in insertion order) ----
  const int*   encoded = (const int*)d_in[0];
  const int*   eidx    = (const int*)d_in[3];
  const int*   batch   = (const int*)d_in[4];
  const float* xnodes  = (const float*)d_in[2];
  auto F = [&](int i){ return (const float*)d_in[i]; };

  // ---- workspace bump allocator ----
  char* wsBase = (char*)d_ws;
  size_t wsOff = 0;
  auto alloc = [&](size_t bytes)->void*{
    size_t o = (wsOff + 255) & ~(size_t)255;
    wsOff = o + bytes;
    return (void*)(wsBase + o);
  };
  auto allocF = [&](size_t n){ return (float*) alloc(n * sizeof(float)); };
  auto allocH = [&](size_t n){ return (half_t*)alloc(n * sizeof(half_t)); };

  // ---- f16 weight staging ----
  half_t* w_s0f_ih = allocH((size_t)kH3*kEMB);
  half_t* w_s0f_hh = allocH((size_t)kH3*kH);
  half_t* w_s0b_ih = allocH((size_t)kH3*kEMB);
  half_t* w_s0b_hh = allocH((size_t)kH3*kH);
  half_t* w_s1f_ih = allocH((size_t)kH3*kH2);
  half_t* w_s1f_hh = allocH((size_t)kH3*kH);
  half_t* w_s1b_ih = allocH((size_t)kH3*kH2);
  half_t* w_s1b_hh = allocH((size_t)kH3*kH);
  half_t* w_e0f_ih = allocH((size_t)kH3*kECFP);
  half_t* w_e0b_ih = allocH((size_t)kH3*kECFP);
  half_t* w_e1f_ih = allocH((size_t)kH3*kH2);
  half_t* w_e1b_ih = allocH((size_t)kH3*kH2);
  half_t *w_sw1[2], *w_sw2[2], *w_sfc[2], *w_efc[2];
  for (int i = 0; i < 2; i++){
    w_sw1[i] = allocH((size_t)kH*kH2);
    w_sw2[i] = allocH((size_t)kOUT*kH);
    w_sfc[i] = allocH((size_t)kOUT*kH2);
    w_efc[i] = allocH((size_t)kOUT*kH2);
  }
  half_t* w_fc1  = allocH((size_t)kOUT*kH2);
  half_t* w_efc2 = allocH((size_t)kOUT*kH2);
  half_t* w_g1t  = allocH((size_t)kGCp*kFXp);
  half_t* w_g2t  = allocH((size_t)kGCp*kGCp);
  half_t* xH     = allocH((size_t)kN*kFXp);
  half_t* ecfpH  = allocH((size_t)kB*kECFP);

  // ---- activation buffers ----
  half_t* embH  = allocH((size_t)kBT*kEMB);
  float*  xwF   = allocF((size_t)kBT*kH3);
  float*  xwB   = allocF((size_t)kBT*kH3);
  float*  gh    = allocF((size_t)kB*kH3);
  float*  hbuf  = allocF((size_t)kB*kH);
  half_t* hbufH = allocH((size_t)kB*kH);
  float*  S0    = allocF((size_t)kBT*kH2);   // layer0 out; later reused as LN(smi)
  half_t* S0h   = allocH((size_t)kBT*kH2);
  float*  S1    = allocF((size_t)kBT*kH2);
  half_t* S1h   = allocH((size_t)kBT*kH2);
  float*  t1    = allocF((size_t)kBT*kH);
  half_t* t1h   = allocH((size_t)kBT*kH);
  float*  sc    = allocF((size_t)kBT*kOUT);
  float*  sfc   = allocF((size_t)kBT*kOUT);
  float*  attacc= allocF((size_t)kBT*kOUT);
  float*  smiatt= allocF((size_t)kB*kOUT);
  float*  epxw  = allocF((size_t)kB*kH3);
  float*  epS0  = allocF((size_t)kB*kH2);
  half_t* epS0h = allocH((size_t)kB*kH2);
  float*  epS1  = allocF((size_t)kB*kH2);
  float*  epL   = allocF((size_t)kB*kH2);
  half_t* epLh  = allocH((size_t)kB*kH2);
  float*  e1    = allocF((size_t)kB*kOUT);
  float*  e2    = allocF((size_t)kB*kOUT);
  float*  e3    = allocF((size_t)kB*kOUT);
  float*  epatt = allocF((size_t)kB*kOUT);
  float*  h1    = allocF((size_t)kN*kGC);
  float*  gacc  = allocF((size_t)kN*kGC);
  float*  g1    = allocF((size_t)kN*kGC);
  half_t* g1h   = allocH((size_t)kN*kGCp);
  float*  h2    = allocF((size_t)kN*kGC);
  float*  g2    = allocF((size_t)kN*kGC);
  float*  ssrc  = allocF((size_t)kN*kHEADS);
  float*  sdst  = allocF((size_t)kN*kHEADS);
  float*  mx    = allocF((size_t)kN*kHEADS);
  float*  den   = allocF((size_t)kN*kHEADS);
  float*  ev    = allocF((size_t)kET*kHEADS);
  float*  ga    = allocF((size_t)kB*kGC);
  float*  cnt   = allocF((size_t)kB);

  // ---- stage weights to f16 (small; fully L2-resident) ----
  auto cvt = [&](int idx, half_t* dst, size_t n){ L1D(cvt_f16_k, n, F(idx), dst, (int)n); };
  cvt(6,  w_s0f_ih, (size_t)kH3*kEMB);  cvt(7,  w_s0f_hh, (size_t)kH3*kH);
  cvt(10, w_s0b_ih, (size_t)kH3*kEMB);  cvt(11, w_s0b_hh, (size_t)kH3*kH);
  cvt(14, w_s1f_ih, (size_t)kH3*kH2);   cvt(15, w_s1f_hh, (size_t)kH3*kH);
  cvt(18, w_s1b_ih, (size_t)kH3*kH2);   cvt(19, w_s1b_hh, (size_t)kH3*kH);
  cvt(22, w_e0f_ih, (size_t)kH3*kECFP); cvt(26, w_e0b_ih, (size_t)kH3*kECFP);
  cvt(30, w_e1f_ih, (size_t)kH3*kH2);   cvt(34, w_e1b_ih, (size_t)kH3*kH2);
  for (int i = 0; i < 2; i++){
    int hb = 42 + 12*i;
    cvt(hb,     w_sw1[i], (size_t)kH*kH2);
    cvt(hb + 2, w_sw2[i], (size_t)kOUT*kH);
    cvt(hb + 8, w_sfc[i], (size_t)kOUT*kH2);
    cvt(hb +10, w_efc[i], (size_t)kOUT*kH2);
  }
  cvt(66, w_fc1,  (size_t)kOUT*kH2);
  cvt(68, w_efc2, (size_t)kOUT*kH2);
  cvt(1,  ecfpH,  (size_t)kB*kECFP);
  L1D(transpose_pad_f16_k, kGCp*kFXp, F(72), kFX, kGC, w_g1t, kFXp, kGCp*kFXp);
  L1D(transpose_pad_f16_k, kGCp*kGCp, F(76), kGC, kGC, w_g2t, kGCp, kGCp*kGCp);
  L1D(cvt_pad_f16_k, kN*kFXp, xnodes, kN, kFX, xH, kFXp, kN*kFXp);

  // ---- embedding + positional encoding ----
  L1D(embed_k, kBT*kEMB, encoded, F(5), embH);

  // ---- BiGRU driver (smi path) ----
  auto run_bigru_layer = [&](const half_t* Xh, int Din,
                             const half_t* wihF, const half_t* whhF,
                             const float* bihF, const float* bhhF,
                             const half_t* wihB, const half_t* whhB,
                             const float* bihB, const float* bhhB,
                             float* Sout, half_t* SoutH){
    gemm(stream, Xh, Din, wihF, Din, xwF, kH3, nullptr, 0, bihF, kBT, kH3, Din, 0);
    gemm(stream, Xh, Din, wihB, Din, xwB, kH3, nullptr, 0, bihB, kBT, kH3, Din, 0);
    // forward
    L1D(fill_f32_k, kB*kH, hbuf, 0.f, kB*kH);
    L1D(fill_f16_k, kB*kH, hbufH, kB*kH);
    for (int t = 0; t < kT; t++){
      gemm(stream, hbufH, kH, whhF, kH, gh, kH3, nullptr, 0, bhhF, kB, kH3, kH, 0);
      L1D(gru_gate_k, kB*kH, xwF, gh, hbuf, hbufH, Sout, SoutH, t, 0);
    }
    // backward
    L1D(fill_f32_k, kB*kH, hbuf, 0.f, kB*kH);
    L1D(fill_f16_k, kB*kH, hbufH, kB*kH);
    for (int s = 0; s < kT; s++){
      int t = kT - 1 - s;
      gemm(stream, hbufH, kH, whhB, kH, gh, kH3, nullptr, 0, bhhB, kB, kH3, kH, 0);
      L1D(gru_gate_k, kB*kH, xwB, gh, hbuf, hbufH, Sout, SoutH, t, kH);
    }
  };

  run_bigru_layer(embH, kEMB, w_s0f_ih, w_s0f_hh, F(8), F(9),
                  w_s0b_ih, w_s0b_hh, F(12), F(13), S0, S0h);
  run_bigru_layer(S0h, kH2, w_s1f_ih, w_s1f_hh, F(16), F(17),
                  w_s1b_ih, w_s1b_hh, F(20), F(21), S1, S1h);
  // LN(smi) -> reuse S0/S0h (layer-0 buffers are dead now)
  layernorm_k<<<kBT, 256, 0, stream>>>(S1, F(38), F(39), S0, S0h, kH2);
  float*  SMI  = S0;
  half_t* SMIh = S0h;

  // ---- ECFP BiGRU (T==1: gh = bhh, h0 = 0) ----
  gemm(stream, ecfpH, kECFP, w_e0f_ih, kECFP, epxw, kH3, nullptr, 0, F(24), kB, kH3, kECFP, 0);
  L1D(gru_single_k, kB*kH, epxw, F(25), epS0, epS0h, 0);
  gemm(stream, ecfpH, kECFP, w_e0b_ih, kECFP, epxw, kH3, nullptr, 0, F(28), kB, kH3, kECFP, 0);
  L1D(gru_single_k, kB*kH, epxw, F(29), epS0, epS0h, kH);
  gemm(stream, epS0h, kH2, w_e1f_ih, kH2, epxw, kH3, nullptr, 0, F(32), kB, kH3, kH2, 0);
  L1D(gru_single_k, kB*kH, epxw, F(33), epS1, (half_t*)nullptr, 0);
  gemm(stream, epS0h, kH2, w_e1b_ih, kH2, epxw, kH3, nullptr, 0, F(36), kB, kH3, kH2, 0);
  L1D(gru_single_k, kB*kH, epxw, F(37), epS1, (half_t*)nullptr, kH);
  layernorm_k<<<kB, 256, 0, stream>>>(epS1, F(40), F(41), epL, epLh, kH2);

  // ---- smi attention heads ----
  L1D(fill_f32_k, kBT*kOUT, attacc, 0.f, kBT*kOUT);
  for (int i = 0; i < kHEADS; i++){
    int hb = 42 + 12*i;
    gemm(stream, SMIh, kH2, w_sw1[i], kH2, t1, kH, t1h, kH, F(hb+1), kBT, kH, kH2, 1); // tanh
    gemm(stream, t1h, kH, w_sw2[i], kH, sc, kOUT, nullptr, 0, F(hb+3), kBT, kOUT, kH, 0);
    L1D(softmax_T_k, kB*kOUT, sc);                                                     // axis=T
    gemm(stream, SMIh, kH2, w_sfc[i], kH2, sfc, kOUT, nullptr, 0, F(hb+9), kBT, kOUT, kH2, 0);
    L1D(mul_acc_k, kBT*kOUT, attacc, sc, sfc, kBT*kOUT);
  }
  gemm(stream, SMIh, kH2, w_fc1, kH2, t1, kOUT, nullptr, 0, F(67), kBT, kOUT, kH2, 0);
  L1D(combine05_k, kBT*kOUT, sc, attacc, t1, kBT*kOUT);   // sc = attacc/HEADS + fc1
  L1D(smi_fc2_k, kB*kOUT, sc, F(70), F(71), smiatt);      // reshape(B,OUT,T) @ fc2

  // ---- ep attention (softmax over length-1 axis == 1.0 -> sw-path drops out) ----
  gemm(stream, epLh, kH2, w_efc[0], kH2, e1, kOUT, nullptr, 0, F(53), kB, kOUT, kH2, 0);
  gemm(stream, epLh, kH2, w_efc[1], kH2, e2, kOUT, nullptr, 0, F(65), kB, kOUT, kH2, 0);
  gemm(stream, epLh, kH2, w_efc2,   kH2, e3, kOUT, nullptr, 0, F(69), kB, kOUT, kH2, 0);
  L1D(ep_combine_k, kB*kOUT, epatt, e1, e2, e3, kB*kOUT);

  // ---- GAT layer 1 ----
  gemm(stream, xH, kFXp, w_g1t, kFXp, h1, kGC, nullptr, 0, nullptr, kN, kGC, kFXp, 0);
  L1D(gat_scores_k, kN*kHEADS, h1, kGC, F(73), F(74), ssrc, sdst);
  L1D(fill_f32_k, kN*kHEADS, mx, -1e30f, kN*kHEADS);
  L1D(fill_f32_k, kN*kHEADS, den, 0.f, kN*kHEADS);
  L1D(gat_edge_max_k, kET*kHEADS, eidx, ssrc, sdst, ev, mx);
  L1D(gat_edge_exp_k, kET*kHEADS, eidx, ev, mx, den);
  L1D(fill_f32_k, kN*kGC, gacc, 0.f, kN*kGC);
  gat_scatter_k<<<kET, 160, 0, stream>>>(eidx, ev, den, h1, gacc);
  L1D(gat_bias_relu_k, kN*kGCp, gacc, F(75), g1, g1h);

  // ---- GAT layer 2 ----
  gemm(stream, g1h, kGCp, w_g2t, kGCp, h2, kGC, nullptr, 0, nullptr, kN, kGC, kGCp, 0);
  L1D(gat_scores_k, kN*kHEADS, h2, kGC, F(77), F(78), ssrc, sdst);
  L1D(fill_f32_k, kN*kHEADS, mx, -1e30f, kN*kHEADS);
  L1D(fill_f32_k, kN*kHEADS, den, 0.f, kN*kHEADS);
  L1D(gat_edge_max_k, kET*kHEADS, eidx, ssrc, sdst, ev, mx);
  L1D(gat_edge_exp_k, kET*kHEADS, eidx, ev, mx, den);
  L1D(fill_f32_k, kN*kGC, gacc, 0.f, kN*kGC);
  gat_scatter_k<<<kET, 160, 0, stream>>>(eidx, ev, den, h2, gacc);
  L1D(gat_bias_relu_k, kN*kGCp, gacc, F(79), g2, (half_t*)nullptr);

  // ---- mean pool per graph ----
  L1D(fill_f32_k, kB*kGC, ga, 0.f, kB*kGC);
  L1D(fill_f32_k, kB, cnt, 0.f, kB);
  L1D(gat_pool_k, kN*kGC, batch, g2, ga, cnt);
  L1D(gat_div_k, kB*kGC, ga, cnt);

  // ---- final linear over concat([smi_att, ep_att, ga]) ----
  L1D(final_k, kB, smiatt, epatt, ga, F(80), F(81), (float*)d_out);
}
